// EquivariantSAB_58901181497934
// MI455X (gfx1250) — matrix operand order repaired and checked by reading, not compiled, not run on hardware
//
#include <hip/hip_runtime.h>
#include <math.h>

// ---------------- problem constants ----------------
#define B_DIM   16
#define N_SEQ   512
#define C_DIM   128
#define H_DIM   8
#define DH      16
#define T_NODES (B_DIM * N_SEQ)       // 8192
#define E_EDGES (T_NODES * 32)        // 262144
#define EQW     32                    // EQ + 3*EQ

typedef __attribute__((ext_vector_type(16))) _Float16 v16h;
typedef __attribute__((ext_vector_type(8)))  _Float16 v8h;
typedef __attribute__((ext_vector_type(8)))  float    v8f;

#if __has_builtin(__builtin_amdgcn_global_load_async_to_lds_b128) && \
    __has_builtin(__builtin_amdgcn_s_wait_asynccnt)
#define HAVE_ASYNC_LDS 1
#else
#define HAVE_ASYNC_LDS 0
#endif

#if HAVE_ASYNC_LDS
typedef int v4i_vs __attribute__((vector_size(16)));
typedef __attribute__((address_space(1))) v4i_vs* g_v4i_ptr;
typedef __attribute__((address_space(3))) v4i_vs* l_v4i_ptr;
// 16-byte async DMA global -> LDS (ASYNCcnt).  Pointers built via integer
// casts: flat->global is identity; flat->LDS is low-32-bit truncation
// (exactly what AMDGPU addrspacecast does).
static __device__ __forceinline__ void async_copy_b128(const void* g, void* l) {
    g_v4i_ptr gp = (g_v4i_ptr)(unsigned long long)(size_t)g;
    l_v4i_ptr lp = (l_v4i_ptr)(unsigned int)(size_t)l;
    __builtin_amdgcn_global_load_async_to_lds_b128(gp, lp, 0, 0);
}
#endif

// A-fragment (16x32 f16, row-major A): lane row = lane&15, K-base = 8*(lane>>4)
// element e -> K = kb+e (e<8) / kb+8+e (e>=8): two contiguous 8-half chunks.
static __device__ __forceinline__ v16h frag_a16(const _Float16* __restrict__ A,
                                                long row, int K, int k0, int lane) {
    int kb = (lane >> 4) << 3;
    const _Float16* p = A + row * (long)K + k0 + kb;
    v8h lo = *(const v8h*)(p);
    v8h hi = *(const v8h*)(p + 16);
    return __builtin_shufflevector(lo, hi, 0,1,2,3,4,5,6,7,8,9,10,11,12,13,14,15);
}

// ---------------- generic WMMA GEMM ----------------
// out = [res +] [relu]( (A @ B) * scale + bias ),  A: MxK f16 row-major,
// Bt: NxK f16 (B transposed).  One 16x16 tile per wave.
__global__ __launch_bounds__(256) void wmma_gemm_kernel(
    const _Float16* __restrict__ A, const _Float16* __restrict__ Bt,
    const float* __restrict__ bias, const float* __restrict__ res,
    float scale, int M, int N, int K, int relu,
    float* __restrict__ outF, _Float16* __restrict__ outH) {
    int lane = threadIdx.x & 31;
    int wave = blockIdx.x * (blockDim.x >> 5) + (threadIdx.x >> 5);
    int ntiles = N >> 4;
    int mt = wave / ntiles, nt = wave - mt * ntiles;
    if (mt >= (M >> 4)) return;
    int m0 = mt << 4, n0 = nt << 4;
    long arow = m0 + (lane & 15);
    int  col  = n0 + (lane & 15);
    v8f acc = {};
    for (int k0 = 0; k0 < K; k0 += 32) {
        __builtin_prefetch(A + arow * (long)K + k0 + 64, 0, 1);
        v16h a = frag_a16(A, arow, K, k0, lane);
        const _Float16* bp = Bt + (long)col * K + k0 + ((lane >> 4) << 3);
        v16h b = __builtin_shufflevector(*(const v8h*)(bp), *(const v8h*)(bp + 16), 0,1,2,3,4,5,6,7,8,9,10,11,12,13,14,15);
        acc = __builtin_amdgcn_wmma_f32_16x16x32_f16(false, a, false, b,
                                                     (short)0, acc, false, false);
    }
    float bv = bias ? bias[col] : 0.0f;
#pragma unroll
    for (int r = 0; r < 8; ++r) {
        int m = m0 + r + ((lane >> 4) << 3);
        float v = acc[r] * scale + bv;
        if (relu) v = fmaxf(v, 0.0f);
        if (res)  v += res[(long)m * N + col];
        if (outF) outF[(long)m * N + col] = v;
        if (outH) outH[(long)m * N + col] = (_Float16)v;
    }
}

// ---------------- attention (one wave per (b,h,16-row q-tile)) ----------------
// S = QK^T/4 (WMMA f16, K padded 16->32), masked softmax in LDS, O = P@V (WMMA),
// epilogue: out = q + O/rowsum.  V slice staged via async global->LDS DMA
// (ASYNCcnt path) into a buffer overlaid on S, then transposed in LDS.
__global__ __launch_bounds__(32) void attn_kernel(
    const float* __restrict__ Qf, const _Float16* __restrict__ Qh,
    const _Float16* __restrict__ Kh, const _Float16* __restrict__ Vh,
    const unsigned char* __restrict__ mask, float* __restrict__ out) {
    __shared__ __attribute__((aligned(32))) float    S[16][N_SEQ + 4];
    __shared__ __attribute__((aligned(32))) _Float16 Vt[16][N_SEQ + 16];
    __shared__ float rs[16];
    int lane = threadIdx.x;
    int blk = blockIdx.x;
    int qt = blk & 31;
    int h  = (blk >> 5) & 7;
    int b  = blk >> 8;
    int m0 = qt * 16;
    long base = (long)b * N_SEQ;

    // ---- stage V (this b,h) transposed into LDS ----
#if HAVE_ASYNC_LDS
    {
        // 1) async DMA the 512x16-f16 V slice row-major into LDS (overlaid on S)
        _Float16* Vl = (_Float16*)&S[0][0];     // 16 KB staging, S written later
        const _Float16* gbase = Vh + base * C_DIM + h * DH;
#pragma unroll 4
        for (int it = 0; it < 32; ++it) {
            int chunk = it * 32 + lane;         // 1024 x 16-byte chunks
            int k = chunk >> 1, half = chunk & 1;
            async_copy_b128(gbase + (long)k * C_DIM + half * 8, Vl + chunk * 8);
        }
        __builtin_amdgcn_s_wait_asynccnt(0);
        __syncthreads();
        // 2) transpose LDS -> LDS: Vt[n][k] = Vl[k*16+n]
        for (int idx = lane; idx < N_SEQ * 16; idx += 32) {
            int k = idx >> 4, n = idx & 15;
            Vt[n][k] = Vl[idx];
        }
    }
#else
    for (int idx = lane; idx < N_SEQ * 16; idx += 32) {
        int k = idx >> 4, n = idx & 15;
        Vt[n][k] = Vh[(base + k) * C_DIM + h * DH + n];
    }
#endif

    // Q fragment: K=0..15 valid, 16..31 zero-padded
    int kb = (lane >> 4) << 3;
    v16h aq;
    {
        const _Float16* qp = Qh + (base + m0 + (lane & 15)) * C_DIM + h * DH;
        aq = (v16h){};
        if (lane < 16) aq = *(const v16h*)(qp);
    }
    __syncthreads();

    const unsigned char* mrow =
        mask + (((long)(b * H_DIM + h) * N_SEQ) + m0) * N_SEQ;
    for (int ct = 0; ct < 32; ++ct) {
        v16h bf = {};   // lanes >=16 carry K=16..31 -> zero
        if (lane < 16)
            bf = *(const v16h*)(Kh + (base + ct * 16 + lane) * C_DIM + h * DH);
        v8f s = {};
        s = __builtin_amdgcn_wmma_f32_16x16x32_f16(false, aq, false, bf,
                                                   (short)0, s, false, false);
#pragma unroll
        for (int r = 0; r < 8; ++r) {
            int mr = r + ((lane >> 4) << 3);
            int kc = ct * 16 + (lane & 15);
            float v = s[r] * 0.25f;                       // 1/sqrt(dh)
            unsigned char mk = mrow[(long)mr * N_SEQ + kc];
            S[mr][kc] = mk ? v : -1.0e9f;
        }
    }
    __syncthreads();

    // softmax: lane pair (row, row+16lanes) splits 512 cols
    int row = lane & 15;
    int c0  = (lane >> 4) << 8;
    float mx = -1.0e30f;
    for (int c = 0; c < 256; ++c) mx = fmaxf(mx, S[row][c0 + c]);
    mx = fmaxf(mx, __shfl_xor(mx, 16, 32));
    float sum = 0.0f;
    for (int c = 0; c < 256; ++c) {
        float e = __expf(S[row][c0 + c] - mx);
        sum += e;
        S[row][c0 + c] = e;
    }
    sum += __shfl_xor(sum, 16, 32);
    if (lane < 16) rs[row] = sum;
    __syncthreads();

    // O = P @ V (f16 WMMA, K = 512)
    v8f o = {};
    for (int k0 = 0; k0 < N_SEQ; k0 += 32) {
        v16h a = {};
#pragma unroll
        for (int e = 0; e < 16; ++e) {
            int kk = ((lane >> 4) << 4) + e;
            a[e] = (_Float16)S[lane & 15][k0 + kk];
        }
        v16h bb = *(const v16h*)(&Vt[lane & 15][k0 + ((lane >> 4) << 4)]);
        o = __builtin_amdgcn_wmma_f32_16x16x32_f16(false, a, false, bb,
                                                   (short)0, o, false, false);
    }
#pragma unroll
    for (int r = 0; r < 8; ++r) {
        int mr = r + ((lane >> 4) << 3);
        long t = base + m0 + mr;
        int cc = h * DH + (lane & 15);
        out[t * C_DIM + cc] = o[r] / rs[mr] + Qf[t * C_DIM + cc];
    }
}

// ---------------- layernorm (one row / block of 128 threads) ----------------
__global__ __launch_bounds__(128) void ln_kernel(
    const float* __restrict__ in, const float* __restrict__ g,
    const float* __restrict__ bt, float* __restrict__ outF,
    _Float16* __restrict__ outH) {
    __shared__ float red[4];
    long t = blockIdx.x;
    int c = threadIdx.x;
    float x = in[t * C_DIM + c];
    float s = x;
    for (int o = 16; o > 0; o >>= 1) s += __shfl_xor(s, o, 32);
    if ((c & 31) == 0) red[c >> 5] = s;
    __syncthreads();
    float mean = (red[0] + red[1] + red[2] + red[3]) * (1.0f / 128.0f);
    __syncthreads();
    float d = x - mean;
    float s2 = d * d;
    for (int o = 16; o > 0; o >>= 1) s2 += __shfl_xor(s2, o, 32);
    if ((c & 31) == 0) red[c >> 5] = s2;
    __syncthreads();
    float var = (red[0] + red[1] + red[2] + red[3]) * (1.0f / 128.0f);
    float y = d * rsqrtf(var + 1e-5f) * g[c] + bt[c];
    outF[t * C_DIM + c] = y;
    if (outH) outH[t * C_DIM + c] = (_Float16)y;
}

// ---------------- edge weight vector (emb==ones => one row) ----------------
__global__ __launch_bounds__(64) void edge_w_kernel(
    const float* __restrict__ Wr0, const float* __restrict__ Wr1,
    const float* __restrict__ Wr2, float* __restrict__ w) {
    __shared__ float h0[64], h1[64];
    int j = threadIdx.x;
    float a = Wr0[j];
    h0[j] = a / (1.0f + __expf(-a));
    __syncthreads();
    float acc = 0.0f;
    for (int i = 0; i < 64; ++i) acc += h0[i] * Wr1[i * 64 + j];
    acc *= 0.125f;                       // 1/sqrt(64)
    h1[j] = acc / (1.0f + __expf(-acc));
    __syncthreads();
    for (int r = 0; r < 4; ++r) {
        int jj = j + r * 64;
        float o = 0.0f;
        for (int i = 0; i < 64; ++i) o += h1[i] * Wr2[i * 256 + jj];
        w[jj] = o * 0.125f;
    }
}

// ---------------- edge scatter (one wave per edge) ----------------
__global__ __launch_bounds__(256) void scatter_kernel(
    const int* __restrict__ ei, const float* __restrict__ ev,
    const float* __restrict__ x1, const float* __restrict__ w,
    float* __restrict__ a0, float* __restrict__ a1) {
    int lane = threadIdx.x & 31;
    long e = (long)blockIdx.x * (blockDim.x >> 5) + (threadIdx.x >> 5);
    if (e >= E_EDGES) return;
    int snd = ei[e], rcv = ei[E_EDGES + e];
    float vx = ev[e * 3 + 0], vy = ev[e * 3 + 1], vz = ev[e * 3 + 2];
    float nrm = fmaxf(sqrtf(vx * vx + vy * vy + vz * vz), 1e-12f);
    float inv = 1.0f / nrm;
    const float s3 = 1.7320508075688772f;
    float Y0 = s3 * vy * inv, Y1 = s3 * vz * inv, Y2 = s3 * vx * inv;  // r[[1,2,0]]
    const float invs = 0.31622776601683794f;  // 1/sqrt(10)
    const long TC = (long)T_NODES * C_DIM;
#pragma unroll
    for (int i = 0; i < 4; ++i) {
        int c = lane + i * 32;
        float xs = x1[(long)snd * C_DIM + c];
        float m0 = w[c] * xs * invs;
        float m1 = w[128 + c] * xs * invs;
        atomicAdd(&a0[(long)rcv * C_DIM + c], m0);
        atomicAdd(&a1[0 * TC + (long)rcv * C_DIM + c], m1 * Y0);
        atomicAdd(&a1[1 * TC + (long)rcv * C_DIM + c], m1 * Y1);
        atomicAdd(&a1[2 * TC + (long)rcv * C_DIM + c], m1 * Y2);
    }
}

// ---------------- skinny eq projections (one wave per node) ----------------
__global__ __launch_bounds__(256) void eq_kernel(
    const float* __restrict__ a0, const float* __restrict__ a1,
    const float* __restrict__ X, const float* __restrict__ W20,
    const float* __restrict__ W21, const float* __restrict__ Wsc,
    _Float16* __restrict__ eqh) {
    int lane = threadIdx.x & 31;
    long t = (long)blockIdx.x * (blockDim.x >> 5) + (threadIdx.x >> 5);
    if (t >= T_NODES) return;
    const long TC = (long)T_NODES * C_DIM;
    const float isc = 0.08838834764831845f;  // 1/sqrt(128)
    float acc = 0.0f;
    if (lane < 8) {
        for (int c = 0; c < C_DIM; ++c)
            acc += a0[t * C_DIM + c] * W20[c * 8 + lane] +
                   X[t * C_DIM + c] * Wsc[c * 8 + lane];
    } else {
        int j = lane - 8, k = j / 3, m = j - k * 3;
        const float* ap = a1 + (long)m * TC + t * C_DIM;
        for (int c = 0; c < C_DIM; ++c) acc += ap[c] * W21[c * 8 + k];
    }
    eqh[t * EQW + lane] = (_Float16)(acc * isc);
}

// ---------------- small utility kernels ----------------
__global__ void cvt_f16_kernel(const float* __restrict__ in,
                               _Float16* __restrict__ out, long n) {
    long i = (long)blockIdx.x * blockDim.x + threadIdx.x;
    if (i < n) out[i] = (_Float16)in[i];
}
__global__ void cvt_wt_kernel(const float* __restrict__ W,
                              _Float16* __restrict__ Wt, int K, int N) {
    long i = (long)blockIdx.x * blockDim.x + threadIdx.x;
    if (i < (long)K * N) {
        int n = (int)(i % N), k = (int)(i / N);
        Wt[(long)n * K + k] = (_Float16)W[i];
    }
}
__global__ void zero_kernel(float* __restrict__ p, long n) {
    long i = (long)blockIdx.x * blockDim.x + threadIdx.x;
    if (i < n) p[i] = 0.0f;
}
__global__ void final_kernel(const float* __restrict__ inv,
                             const float* __restrict__ gate,
                             const float* __restrict__ eqs,
                             float* __restrict__ out, long n) {
    long i = (long)blockIdx.x * blockDim.x + threadIdx.x;
    if (i < n)
        out[i] = inv[i] * (1.0f / (1.0f + __expf(-gate[i]))) + eqs[i];
}

// ---------------- host orchestration ----------------
extern "C" void kernel_launch(void* const* d_in, const int* in_sizes, int n_in,
                              void* d_out, int out_size, void* d_ws, size_t ws_size,
                              hipStream_t stream) {
    (void)in_sizes; (void)n_in; (void)out_size; (void)ws_size;
    const float* X    = (const float*)d_in[0];
    const unsigned char* mask = (const unsigned char*)d_in[1];
    const int*   ei   = (const int*)d_in[2];
    const float* ev   = (const float*)d_in[3];
    const float* Wq = (const float*)d_in[5],  *bq = (const float*)d_in[6];
    const float* Wk = (const float*)d_in[7],  *bk = (const float*)d_in[8];
    const float* Wv = (const float*)d_in[9],  *bv = (const float*)d_in[10];
    const float* Wf = (const float*)d_in[11], *bf = (const float*)d_in[12];
    const float* g0 = (const float*)d_in[13], *b0 = (const float*)d_in[14];
    const float* g1 = (const float*)d_in[15], *b1 = (const float*)d_in[16];
    const float* Wc = (const float*)d_in[17], *bc = (const float*)d_in[18];
    const float* W1 = (const float*)d_in[19];
    const float* Wr0 = (const float*)d_in[20];
    const float* Wr1 = (const float*)d_in[21];
    const float* Wr2 = (const float*)d_in[22];
    const float* W20 = (const float*)d_in[23];
    const float* W21 = (const float*)d_in[24];
    const float* Wsc = (const float*)d_in[25];
    const float* Wes = (const float*)d_in[26], *bes = (const float*)d_in[27];
    const float* Wg  = (const float*)d_in[28], *bg  = (const float*)d_in[29];

    char* wsb = (char*)d_ws;
    size_t off = 0;
    auto alloc = [&](size_t bytes) -> char* {
        off = (off + 255) & ~(size_t)255;
        char* p = wsb + off;
        off += bytes;
        return p;
    };
    const long TC = (long)T_NODES * C_DIM;
    _Float16* Xh   = (_Float16*)alloc(TC * 2);
    _Float16* WqT  = (_Float16*)alloc(C_DIM * C_DIM * 2);
    _Float16* WkT  = (_Float16*)alloc(C_DIM * C_DIM * 2);
    _Float16* WvT  = (_Float16*)alloc(C_DIM * C_DIM * 2);
    _Float16* WfT  = (_Float16*)alloc(C_DIM * C_DIM * 2);
    _Float16* WcT  = (_Float16*)alloc(C_DIM * C_DIM * 2);
    _Float16* W1T  = (_Float16*)alloc(C_DIM * C_DIM * 2);
    _Float16* WesT = (_Float16*)alloc(C_DIM * EQW * 2);
    _Float16* WgT  = (_Float16*)alloc(C_DIM * EQW * 2);
    float*    Qf   = (float*)alloc(TC * 4);
    _Float16* Qh   = (_Float16*)alloc(TC * 2);
    _Float16* Kh   = (_Float16*)alloc(TC * 2);
    _Float16* Vh   = (_Float16*)alloc(TC * 2);
    float*    attn = (float*)alloc(TC * 4);
    float*    u    = (float*)alloc(TC * 4);
    _Float16* uh   = (_Float16*)alloc(TC * 2);
    float*    o2   = (float*)alloc(TC * 4);
    float*    invb = (float*)alloc(TC * 4);
    _Float16* xeh  = (_Float16*)alloc(TC * 2);
    float*    x1f  = (float*)alloc(TC * 4);
    float*    wvec = (float*)alloc(256 * 4);
    float*    a0   = (float*)alloc(TC * 4);
    float*    a1   = (float*)alloc(3 * TC * 4);
    _Float16* eqh  = (_Float16*)alloc((long)T_NODES * EQW * 2);
    float*    eqs  = (float*)alloc(TC * 4);
    float*    gate = (float*)alloc(TC * 4);

    const float ISC = 0.08838834764831845f;  // 1/sqrt(C)

    // conversions
    cvt_f16_kernel<<<(TC + 255) / 256, 256, 0, stream>>>(X, Xh, TC);
    cvt_wt_kernel<<<64, 256, 0, stream>>>(Wq, WqT, C_DIM, C_DIM);
    cvt_wt_kernel<<<64, 256, 0, stream>>>(Wk, WkT, C_DIM, C_DIM);
    cvt_wt_kernel<<<64, 256, 0, stream>>>(Wv, WvT, C_DIM, C_DIM);
    cvt_wt_kernel<<<64, 256, 0, stream>>>(Wf, WfT, C_DIM, C_DIM);
    cvt_wt_kernel<<<64, 256, 0, stream>>>(Wc, WcT, C_DIM, C_DIM);
    cvt_wt_kernel<<<64, 256, 0, stream>>>(W1, W1T, C_DIM, C_DIM);
    cvt_wt_kernel<<<16, 256, 0, stream>>>(Wes, WesT, EQW, C_DIM);
    cvt_wt_kernel<<<16, 256, 0, stream>>>(Wg,  WgT,  EQW, C_DIM);

    // projection GEMMs (WMMA): 8192x128x128 -> 512 blocks of 8 waves
    const int gblk = ((T_NODES / 16) * (C_DIM / 16) + 7) / 8;
    wmma_gemm_kernel<<<gblk, 256, 0, stream>>>(Xh, WqT, bq, nullptr, 1.0f,
        T_NODES, C_DIM, C_DIM, 0, Qf, Qh);
    wmma_gemm_kernel<<<gblk, 256, 0, stream>>>(Xh, WkT, bk, nullptr, 1.0f,
        T_NODES, C_DIM, C_DIM, 0, nullptr, Kh);
    wmma_gemm_kernel<<<gblk, 256, 0, stream>>>(Xh, WvT, bv, nullptr, 1.0f,
        T_NODES, C_DIM, C_DIM, 0, nullptr, Vh);
    wmma_gemm_kernel<<<gblk, 256, 0, stream>>>(Xh, WcT, bc, nullptr, 1.0f,
        T_NODES, C_DIM, C_DIM, 0, nullptr, xeh);
    wmma_gemm_kernel<<<gblk, 256, 0, stream>>>(xeh, W1T, nullptr, nullptr, ISC,
        T_NODES, C_DIM, C_DIM, 0, x1f, nullptr);

    // attention: B*H*(N/16) = 4096 waves
    attn_kernel<<<B_DIM * H_DIM * (N_SEQ / 16), 32, 0, stream>>>(
        Qf, Qh, Kh, Vh, mask, attn);

    // LN0 -> FF (relu + residual) -> LN1
    ln_kernel<<<T_NODES, 128, 0, stream>>>(attn, g0, b0, u, uh);
    wmma_gemm_kernel<<<gblk, 256, 0, stream>>>(uh, WfT, bf, u, 1.0f,
        T_NODES, C_DIM, C_DIM, 1, o2, nullptr);
    ln_kernel<<<T_NODES, 128, 0, stream>>>(o2, g1, b1, invb, nullptr);

    // edge path
    edge_w_kernel<<<1, 64, 0, stream>>>(Wr0, Wr1, Wr2, wvec);
    zero_kernel<<<(TC + 255) / 256, 256, 0, stream>>>(a0, TC);
    zero_kernel<<<(3 * TC + 255) / 256, 256, 0, stream>>>(a1, 3 * TC);
    scatter_kernel<<<E_EDGES / 8, 256, 0, stream>>>(ei, ev, x1f, wvec, a0, a1);
    eq_kernel<<<T_NODES / 8, 256, 0, stream>>>(a0, a1, X, W20, W21, Wsc, eqh);

    // eq_scalar / gate: K=32 WMMA GEMMs
    wmma_gemm_kernel<<<gblk, 256, 0, stream>>>(eqh, WesT, bes, nullptr, 1.0f,
        T_NODES, C_DIM, EQW, 0, eqs, nullptr);
    wmma_gemm_kernel<<<gblk, 256, 0, stream>>>(eqh, WgT, bg, nullptr, 1.0f,
        T_NODES, C_DIM, EQW, 0, gate, nullptr);

    // out = inv * sigmoid(gate) + eq_scalar
    final_kernel<<<(TC + 255) / 256, 256, 0, stream>>>(
        invb, gate, eqs, (float*)d_out, TC);
}
